// CoAttentionAndCombine_1168231104741
// MI455X (gfx1250) — compile-verified, matrix operand order checked by reading, not run on hardware
//
#include <hip/hip_runtime.h>
#include <hip/hip_bf16.h>

// Problem constants (match the reference)
#define B_ 16
#define L_ 1024
#define D_ 768
#define MASK_SCALE 10000.0f

typedef __bf16 bf16;
typedef __attribute__((ext_vector_type(16))) __bf16 v16bf;
typedef __attribute__((ext_vector_type(8)))  float  v8f;
typedef __attribute__((ext_vector_type(4)))  unsigned int u32x4;
typedef __attribute__((ext_vector_type(8)))  unsigned int u32x8;

static __device__ inline v8f zero_v8f() {
  v8f z;
#pragma unroll
  for (int i = 0; i < 8; ++i) z[i] = 0.0f;
  return z;
}

static __device__ inline v8f wmma_bf16(v16bf a, v16bf b, v8f c) {
  // D = A(16x32 bf16) * B(32x16 bf16) + C(16x16 f32)
  return __builtin_amdgcn_wmma_f32_16x16x32_bf16(
      /*neg_a=*/false, a, /*neg_b=*/false, b,
      /*c_mod=*/(short)0, c, /*reuse_a=*/false, /*reuse_b=*/false);
}

// 16-bit A-matrix 16x32 fragment (ISA 7.12.2): lanes 0-15 hold row M=lane,
// K = k0+0..7 and k0+16..23; lanes 16-31 hold K = k0+8..15 and k0+24..31.
// Per lane: two contiguous 16-byte chunks -> global_load_b128 pair.
// Also used for B operands: with the B-source stored transposed (row-major
// [n][k]), each lane's column chunk is contiguous as well.
static __device__ inline v16bf load_rm_frag(const bf16* __restrict__ base_row0,
                                            int ld, int k0, unsigned lane) {
  const bf16* p = base_row0 + (size_t)(lane & 15u) * ld + k0 + ((lane >> 4) << 3);
  union { uint4 u[2]; v16bf v; } c;
  c.u[0] = *(const uint4*)(p);
  c.u[1] = *(const uint4*)(p + 16);
  return c.v;
}

// ---------------------------------------------------------------------------
// Tensor Data Mover: 2D fp32 tile (tile_x contiguous elems per row, tile_y
// rows, row stride stride_elems) from global into dense row-major LDS.
// D# per ISA ch.8: group0 {count, lds_addr, global_addr, type=2},
// group1 {data_size=4B, tensor dims, tile dims, dim0 stride}.
// ---------------------------------------------------------------------------
static __device__ inline void tdm_load_2d_f32(unsigned lds_off, const void* gaddr,
                                              unsigned tile_x, unsigned tile_y,
                                              unsigned stride_elems,
                                              unsigned tdim0, unsigned tdim1) {
  unsigned long long ga = (unsigned long long)(size_t)gaddr;
  u32x4 g0;
  g0[0] = 1u;                                   // count=1, user descriptor
  g0[1] = lds_off;                              // lds_addr (bytes)
  g0[2] = (unsigned)(ga & 0xFFFFFFFFu);         // global_addr[31:0]
  g0[3] = (unsigned)((ga >> 32) & 0x01FFFFFFu)  // global_addr[56:32]
          | (2u << 30);                         // type = 2 ("image")
  u32x8 g1;
  g1[0] = 2u << 16;                             // wg_mask=0, data_size=4B
  g1[1] = (tdim0 & 0xFFFFu) << 16;              // tensor_dim0[15:0] @ bits 63:48
  g1[2] = ((tdim0 >> 16) & 0xFFFFu)             // tensor_dim0[31:16]
          | ((tdim1 & 0xFFFFu) << 16);          // tensor_dim1[15:0]
  g1[3] = ((tdim1 >> 16) & 0xFFFFu)             // tensor_dim1[31:16]
          | (tile_x << 16);                     // tile_dim0
  g1[4] = tile_y & 0xFFFFu;                     // tile_dim1 (tile_dim2 = 0)
  g1[5] = stride_elems;                         // tensor_dim0_stride[31:0]
  g1[6] = 0;                                    // stride hi, dim1_stride lo
  g1[7] = 0;
  asm volatile("tensor_load_to_lds %0, %1" :: "s"(g0), "s"(g1) : "memory");
}

// ---------------------------------------------------------------------------
// 1) fused fp32 -> bf16 convert + transpose (writes both layouts) via LDS tile
//    in: [z][R][C] f32 ; out_rm: [z][R][C] bf16 ; out_tr: [z][C][R] bf16
// ---------------------------------------------------------------------------
__global__ void __launch_bounds__(256) cvt_t_kernel(const float* __restrict__ in,
                                                    bf16* __restrict__ out_rm,
                                                    bf16* __restrict__ out_tr,
                                                    int R, int C) {
  __shared__ bf16 st[32][33];
  const int z = blockIdx.z;
  const float* inb = in + (size_t)z * R * C;
  bf16* orm = out_rm + (size_t)z * R * C;
  bf16* otr = out_tr + (size_t)z * R * C;
  const int c0 = blockIdx.x * 32, r0 = blockIdx.y * 32;
  const int x = threadIdx.x, y = threadIdx.y;
#pragma unroll
  for (int i = 0; i < 32; i += 8) {
    int r = r0 + y + i;
    bf16 h = (bf16)inb[(size_t)r * C + c0 + x];
    orm[(size_t)r * C + c0 + x] = h;
    st[y + i][x] = h;
  }
  __syncthreads();
#pragma unroll
  for (int i = 0; i < 32; i += 8) {
    int c = c0 + y + i;
    otr[(size_t)c * R + r0 + x] = st[x][y + i];
  }
}

// ---------------------------------------------------------------------------
// 2) Y = X (BL x D) * W (D x D). B frags from pre-transposed Wt (contiguous).
//    Block 64 x 128, 8 waves, 4 WMMAs per wave per K-step.
// ---------------------------------------------------------------------------
__global__ void __launch_bounds__(256) ymm_kernel(const bf16* __restrict__ X,
                                                  const bf16* __restrict__ Wt,
                                                  bf16* __restrict__ Y) {
  const unsigned lane = threadIdx.x & 31u;
  const unsigned wave = threadIdx.x >> 5;
  const int row0 = blockIdx.x * 64 + (int)(wave & 3u) * 16;
  const int n0   = blockIdx.y * 128 + (int)(wave >> 2) * 64;
  const bf16* Arow = X + (size_t)row0 * D_;

  v8f acc[4] = {zero_v8f(), zero_v8f(), zero_v8f(), zero_v8f()};
  for (int k0 = 0; k0 < D_; k0 += 32) {
    if (k0 + 32 < D_)
      __builtin_prefetch(Arow + (size_t)(lane & 15u) * D_ + k0 + 32, 0, 1);
    v16bf a = load_rm_frag(Arow, D_, k0, lane);
#pragma unroll
    for (int s = 0; s < 4; ++s) {
      v16bf bb = load_rm_frag(Wt + (size_t)(n0 + 16 * s) * D_, D_, k0, lane);
      acc[s] = wmma_bf16(a, bb, acc[s]);
    }
  }
  const int half = (int)(lane >> 4);
  const int nl   = (int)(lane & 15u);
#pragma unroll
  for (int s = 0; s < 4; ++s)
#pragma unroll
    for (int r = 0; r < 8; ++r) {
      int m = row0 + half * 8 + r;
      Y[(size_t)m * D_ + n0 + 16 * s + nl] = (bf16)acc[s][r];
    }
}

// ---------------------------------------------------------------------------
// 3) sim[b,i,j] = Y[b,i,:] . x2[b,j,:] + (m1*m2-1)*1e4
//    B = x2^T: row-major x2 gives contiguous column chunks. Block 64 x 128.
// ---------------------------------------------------------------------------
__global__ void __launch_bounds__(256) sim_kernel(const bf16* __restrict__ Y,
                                                  const bf16* __restrict__ X2,
                                                  const float* __restrict__ m1,
                                                  const float* __restrict__ m2,
                                                  float* __restrict__ sim) {
  const unsigned lane = threadIdx.x & 31u;
  const unsigned wave = threadIdx.x >> 5;
  const int b   = blockIdx.z;
  const int i0  = blockIdx.x * 64 + (int)(wave & 3u) * 16;
  const int j0w = blockIdx.y * 128 + (int)(wave >> 2) * 64;
  const bf16* Yb  = Y  + ((size_t)b * L_ + i0) * D_;
  const bf16* X2b = X2 + ((size_t)b * L_ + j0w) * D_;

  v8f acc[4] = {zero_v8f(), zero_v8f(), zero_v8f(), zero_v8f()};
  for (int k0 = 0; k0 < D_; k0 += 32) {
    if (k0 + 32 < D_)
      __builtin_prefetch(Yb + (size_t)(lane & 15u) * D_ + k0 + 32, 0, 1);
    v16bf a = load_rm_frag(Yb, D_, k0, lane);
#pragma unroll
    for (int s = 0; s < 4; ++s) {
      v16bf bb = load_rm_frag(X2b + (size_t)(16 * s) * D_, D_, k0, lane);
      acc[s] = wmma_bf16(a, bb, acc[s]);
    }
  }
  const int half = (int)(lane >> 4);
  const int nl   = (int)(lane & 15u);
  float* simb = sim + (size_t)b * L_ * L_;
#pragma unroll
  for (int s = 0; s < 4; ++s) {
    const float m2v = m2[b * L_ + j0w + 16 * s + nl];
#pragma unroll
    for (int r = 0; r < 8; ++r) {
      int i = i0 + half * 8 + r;
      float mb = m1[b * L_ + i];
      simb[(size_t)i * L_ + j0w + 16 * s + nl] =
          acc[s][r] + (mb * m2v - 1.0f) * MASK_SCALE;
    }
  }
}

// ---------------------------------------------------------------------------
// 4) softmax stats: per output-row (block), max and sum(exp(v-max)) over L
//    elements at base + j*colstride.
// ---------------------------------------------------------------------------
__global__ void __launch_bounds__(256) stats_kernel(const float* __restrict__ sim,
                                                    int rowstride, int colstride,
                                                    float* __restrict__ omax,
                                                    float* __restrict__ osum) {
  __shared__ float red[256];
  const int row = blockIdx.x;  // 0 .. B*L-1
  const int b = row / L_;
  const int r = row % L_;
  const float* base = sim + (size_t)b * L_ * L_ + (size_t)r * rowstride;
  const int tid = threadIdx.x;

  float v[4];
  float mx = -3.0e38f;
#pragma unroll
  for (int t = 0; t < 4; ++t) {
    int j = tid + t * 256;
    v[t] = base[(size_t)j * colstride];
    mx = fmaxf(mx, v[t]);
  }
  red[tid] = mx;
  __syncthreads();
  for (int s = 128; s > 0; s >>= 1) {
    if (tid < s) red[tid] = fmaxf(red[tid], red[tid + s]);
    __syncthreads();
  }
  mx = red[0];
  __syncthreads();
  float sum = 0.0f;
#pragma unroll
  for (int t = 0; t < 4; ++t) sum += __expf(v[t] - mx);
  red[tid] = sum;
  __syncthreads();
  for (int s = 128; s > 0; s >>= 1) {
    if (tid < s) red[tid] += red[tid + s];
    __syncthreads();
  }
  if (tid == 0) { omax[row] = mx; osum[row] = red[0]; }
}

// ---------------------------------------------------------------------------
// 5a) x1_tilde = softmax_rows(sim) * x2. A = probabilities built on the fly
//     from contiguous sim rows; B frags from pre-transposed x2 (contiguous).
//     Block 64 m x 128 d, 4 WMMAs/wave/step, K loop over j (L).
// ---------------------------------------------------------------------------
__global__ void __launch_bounds__(256) tilde_row_kernel(const float* __restrict__ sim,
                                                        const float* __restrict__ Mx,
                                                        const float* __restrict__ Sm,
                                                        const bf16* __restrict__ Vt,
                                                        float* __restrict__ out) {
  const unsigned lane = threadIdx.x & 31u;
  const unsigned wave = threadIdx.x >> 5;
  const int b   = blockIdx.z;
  const int m0  = blockIdx.x * 64 + (int)(wave & 3u) * 16;
  const int d0w = blockIdx.y * 128 + (int)(wave >> 2) * 64;
  const float* simb = sim + (size_t)b * L_ * L_;
  const bf16*  Vtb  = Vt + (size_t)b * L_ * D_;   // [D][L] layout

  const int   m  = m0 + (int)(lane & 15u);
  const float mx = Mx[b * L_ + m];
  const float is = 1.0f / Sm[b * L_ + m];
  const float* prow = simb + (size_t)m * L_;

  v8f acc[4] = {zero_v8f(), zero_v8f(), zero_v8f(), zero_v8f()};
  for (int k0 = 0; k0 < L_; k0 += 32) {
    const int kb = k0 + (int)((lane >> 4) << 3);
    if (k0 + 32 < L_) __builtin_prefetch(prow + kb + 32, 0, 1);
    float4 f0 = *(const float4*)(prow + kb);
    float4 f1 = *(const float4*)(prow + kb + 4);
    float4 f2 = *(const float4*)(prow + kb + 16);
    float4 f3 = *(const float4*)(prow + kb + 20);
    v16bf a;
    a[0]  = (bf16)(__expf(f0.x - mx) * is);
    a[1]  = (bf16)(__expf(f0.y - mx) * is);
    a[2]  = (bf16)(__expf(f0.z - mx) * is);
    a[3]  = (bf16)(__expf(f0.w - mx) * is);
    a[4]  = (bf16)(__expf(f1.x - mx) * is);
    a[5]  = (bf16)(__expf(f1.y - mx) * is);
    a[6]  = (bf16)(__expf(f1.z - mx) * is);
    a[7]  = (bf16)(__expf(f1.w - mx) * is);
    a[8]  = (bf16)(__expf(f2.x - mx) * is);
    a[9]  = (bf16)(__expf(f2.y - mx) * is);
    a[10] = (bf16)(__expf(f2.z - mx) * is);
    a[11] = (bf16)(__expf(f2.w - mx) * is);
    a[12] = (bf16)(__expf(f3.x - mx) * is);
    a[13] = (bf16)(__expf(f3.y - mx) * is);
    a[14] = (bf16)(__expf(f3.z - mx) * is);
    a[15] = (bf16)(__expf(f3.w - mx) * is);
#pragma unroll
    for (int s = 0; s < 4; ++s) {
      v16bf bb = load_rm_frag(Vtb + (size_t)(d0w + 16 * s) * L_, L_, k0, lane);
      acc[s] = wmma_bf16(a, bb, acc[s]);
    }
  }
  const int half = (int)(lane >> 4);
  const int nl   = (int)(lane & 15u);
#pragma unroll
  for (int s = 0; s < 4; ++s)
#pragma unroll
    for (int r = 0; r < 8; ++r) {
      int mm = m0 + half * 8 + r;
      out[(size_t)(b * L_ + mm) * (4 * D_) + D_ + d0w + 16 * s + nl] = acc[s][r];
    }
}

// ---------------------------------------------------------------------------
// 5b) x2_tilde = softmax_cols(sim)^T * x1. The 32x64 fp32 sim K-tile is DMA'd
//     into LDS with the Tensor Data Mover (strided rows -> dense LDS), then
//     column-read from LDS to build the probability A-fragment.
// ---------------------------------------------------------------------------
__global__ void __launch_bounds__(256) tilde_col_kernel(const float* __restrict__ sim,
                                                        const float* __restrict__ Mx,
                                                        const float* __restrict__ Sm,
                                                        const bf16* __restrict__ Vt,
                                                        float* __restrict__ out) {
  __shared__ __align__(16) float stile[32 * 64];  // [k][j] dense
  const unsigned lane = threadIdx.x & 31u;
  const unsigned wave = threadIdx.x >> 5;
  const int b   = blockIdx.z;
  const int jb  = blockIdx.x * 64;                       // block j (row) base
  const int m0  = jb + (int)(wave & 3u) * 16;
  const int d0w = blockIdx.y * 128 + (int)(wave >> 2) * 64;
  const float* simb = sim + (size_t)b * L_ * L_;
  const bf16*  Vtb  = Vt + (size_t)b * L_ * D_;          // [D][L] layout

  const int   m     = m0 + (int)(lane & 15u);            // output row = j
  const int   m_rel = (int)(wave & 3u) * 16 + (int)(lane & 15u);
  const float mx = Mx[b * L_ + m];
  const float is = 1.0f / Sm[b * L_ + m];

  v8f acc[4] = {zero_v8f(), zero_v8f(), zero_v8f(), zero_v8f()};
  for (int k0 = 0; k0 < L_; k0 += 32) {
    __syncthreads();
    if (wave == 0) {
      tdm_load_2d_f32((unsigned)(size_t)&stile[0],
                      simb + (size_t)k0 * L_ + jb,
                      /*tile_x=*/64, /*tile_y=*/32,
                      /*stride=*/L_, /*tdim0=*/L_, /*tdim1=*/L_);
      __builtin_amdgcn_s_wait_tensorcnt(0);
    }
    __syncthreads();
    const int kb = (int)((lane >> 4) << 3);
    v16bf a;
#pragma unroll
    for (int t = 0; t < 8; ++t) {
      float s0 = stile[(kb + t) * 64 + m_rel];
      float s1 = stile[(kb + 16 + t) * 64 + m_rel];
      a[t]     = (bf16)(__expf(s0 - mx) * is);
      a[8 + t] = (bf16)(__expf(s1 - mx) * is);
    }
#pragma unroll
    for (int s = 0; s < 4; ++s) {
      v16bf bb = load_rm_frag(Vtb + (size_t)(d0w + 16 * s) * L_, L_, k0, lane);
      acc[s] = wmma_bf16(a, bb, acc[s]);
    }
  }
  const int half = (int)(lane >> 4);
  const int nl   = (int)(lane & 15u);
#pragma unroll
  for (int s = 0; s < 4; ++s)
#pragma unroll
    for (int r = 0; r < 8; ++r) {
      int mm = m0 + half * 8 + r;
      out[(size_t)(b * L_ + mm) * (4 * D_) + D_ + d0w + 16 * s + nl] = acc[s][r];
    }
}

// ---------------------------------------------------------------------------
// 6) combine: out = [x, xt, |x-xt|, x*xt]; xt already sits in quarter 1.
// ---------------------------------------------------------------------------
__global__ void __launch_bounds__(256) combine_kernel(const float* __restrict__ x,
                                                      float* __restrict__ out,
                                                      int totalVec) {
  int idx = blockIdx.x * 256 + threadIdx.x;  // float4 index
  if (idx >= totalVec) return;
  const int dv  = idx % (D_ / 4);
  const int row = idx / (D_ / 4);
  const float4 a = ((const float4*)x)[idx];
  float4* o = (float4*)(out + (size_t)row * (4 * D_));
  const float4 t = o[D_ / 4 + dv];
  float4 ab, mu;
  ab.x = fabsf(a.x - t.x); ab.y = fabsf(a.y - t.y);
  ab.z = fabsf(a.z - t.z); ab.w = fabsf(a.w - t.w);
  mu.x = a.x * t.x; mu.y = a.y * t.y; mu.z = a.z * t.z; mu.w = a.w * t.w;
  o[dv]              = a;
  o[2 * D_ / 4 + dv] = ab;
  o[3 * D_ / 4 + dv] = mu;
}

// ---------------------------------------------------------------------------
extern "C" void kernel_launch(void* const* d_in, const int* in_sizes, int n_in,
                              void* d_out, int out_size, void* d_ws, size_t ws_size,
                              hipStream_t stream) {
  (void)in_sizes; (void)n_in; (void)out_size; (void)ws_size;
  const float* x1 = (const float*)d_in[0];
  const float* m1 = (const float*)d_in[1];
  const float* x2 = (const float*)d_in[2];
  const float* m2 = (const float*)d_in[3];
  const float* W  = (const float*)d_in[4];
  float* out = (float*)d_out;

  const size_t nXD = (size_t)B_ * L_ * D_;   // 12,582,912
  const size_t nWW = (size_t)D_ * D_;        // 589,824
  const size_t nBL = (size_t)B_ * L_;        // 16,384
  const size_t nSS = (size_t)B_ * L_ * L_;   // 16,777,216

  // Workspace carve-up (~196 MB); sim (67 MB) + bf16 operands stay L2-hot.
  char* ws = (char*)d_ws;
  size_t off = 0;
  auto carve = [&](size_t bytes) -> char* {
    char* p = ws + off;
    off += (bytes + 255) & ~(size_t)255;
    return p;
  };
  bf16*  x1b  = (bf16*)carve(nXD * sizeof(bf16));
  bf16*  x2b  = (bf16*)carve(nXD * sizeof(bf16));
  bf16*  x1bT = (bf16*)carve(nXD * sizeof(bf16));  // [b][d][l]
  bf16*  x2bT = (bf16*)carve(nXD * sizeof(bf16));  // [b][d][l]
  bf16*  wb   = (bf16*)carve(nWW * sizeof(bf16));
  bf16*  wbT  = (bf16*)carve(nWW * sizeof(bf16));  // [e][k]
  bf16*  yb   = (bf16*)carve(nXD * sizeof(bf16));
  float* sim  = (float*)carve(nSS * sizeof(float));
  float* rmax = (float*)carve(nBL * sizeof(float));
  float* rsum = (float*)carve(nBL * sizeof(float));
  float* cmax = (float*)carve(nBL * sizeof(float));
  float* csum = (float*)carve(nBL * sizeof(float));

  float* out1 = out;
  float* out2 = out + nXD * 4;  // second tuple element, concatenated flat

  dim3 blk(256);
  dim3 blkT(32, 8);

  // 1) bf16 conversions + transposes (fused)
  cvt_t_kernel<<<dim3(D_ / 32, L_ / 32, B_), blkT, 0, stream>>>(x1, x1b, x1bT, L_, D_);
  cvt_t_kernel<<<dim3(D_ / 32, L_ / 32, B_), blkT, 0, stream>>>(x2, x2b, x2bT, L_, D_);
  cvt_t_kernel<<<dim3(D_ / 32, D_ / 32, 1), blkT, 0, stream>>>(W, wb, wbT, D_, D_);

  // 2) Y = x1 * W
  ymm_kernel<<<dim3((B_ * L_) / 64, D_ / 128), blk, 0, stream>>>(x1b, wbT, yb);

  // 3) sim = Y * x2^T + mask_bias
  sim_kernel<<<dim3(L_ / 64, L_ / 128, B_), blk, 0, stream>>>(yb, x2b, m1, m2, sim);

  // 4) softmax stats over rows (axis j) and columns (axis i)
  stats_kernel<<<B_ * L_, blk, 0, stream>>>(sim, L_, 1, rmax, rsum);
  stats_kernel<<<B_ * L_, blk, 0, stream>>>(sim, 1, L_, cmax, csum);

  // 5) x1_tilde and x2_tilde straight into the output quarters
  tilde_row_kernel<<<dim3(L_ / 64, D_ / 128, B_), blk, 0, stream>>>(
      sim, rmax, rsum, x2bT, out1);
  tilde_col_kernel<<<dim3(L_ / 64, D_ / 128, B_), blk, 0, stream>>>(
      sim, cmax, csum, x1bT, out2);

  // 6) combine quarters (float4 streaming)
  combine_kernel<<<(unsigned)((nXD / 4 + 255) / 256), blk, 0, stream>>>(
      x1, out1, (int)(nXD / 4));
  combine_kernel<<<(unsigned)((nXD / 4 + 255) / 256), blk, 0, stream>>>(
      x2, out2, (int)(nXD / 4));
}
